// TidalTransformer_14044543058093
// MI455X (gfx1250) — matrix operand order repaired and checked by reading, not compiled, not run on hardware
//
#include <hip/hip_runtime.h>
#include <hip/hip_bf16.h>
#include <math.h>
#include <stdint.h>

// ---------------------------------------------------------------------------
// CDNA5 async global->LDS support via inline asm (bypasses clang builtin
// signature differences between toolchains). Sync fallback for host pass.
// ---------------------------------------------------------------------------
#if defined(__HIP_DEVICE_COMPILE__) && defined(__gfx1250__)
#define HAVE_ASYNC 1
#else
#define HAVE_ASYNC 0
#endif

#if HAVE_ASYNC
#define WAIT_ASYNC(n) asm volatile("s_wait_asynccnt %0" ::"i"(n) : "memory")
#else
#define WAIT_ASYNC(n) ((void)0)
#endif

// copy 8 bytes global -> LDS (async DMA when available)
__device__ __forceinline__ void stage8(void* lds, const void* g) {
#if HAVE_ASYNC
    unsigned laddr = (unsigned)(uintptr_t)lds;   // LDS aperture: low 32 bits = LDS offset
    asm volatile("global_load_async_to_lds_b64 %0, %1, off"
                 ::"v"(laddr), "v"(g)
                 : "memory");
#else
    *(unsigned long long*)lds = *(const unsigned long long*)g;
#endif
}

// ---------------------------------------------------------------------------
// Types / helpers
// ---------------------------------------------------------------------------
typedef __attribute__((ext_vector_type(16))) __bf16 v16bf;
typedef __attribute__((ext_vector_type(8)))  float  v8f;
typedef __attribute__((ext_vector_type(8)))  unsigned int v8u;

union Frag { v8u u; v16bf b; };

#define WMMA_BF16(A_, B_, C_) \
    __builtin_amdgcn_wmma_f32_16x16x32_bf16(false, (A_), false, (B_), (short)0, (C_), false, false)

__device__ __forceinline__ v8f v8f_zero() {
    v8f z = {0.f,0.f,0.f,0.f,0.f,0.f,0.f,0.f};
    return z;
}
__device__ __forceinline__ unsigned short bf16of(float a) {
    unsigned ua = __float_as_uint(a);
    ua += 0x7FFFu + ((ua >> 16) & 1u);
    return (unsigned short)(ua >> 16);
}

// ---------------------------------------------------------------------------
// Weight convert + transpose: W fp32 [K,N]  ->  Wt bf16 [N,K]
// ---------------------------------------------------------------------------
__global__ __launch_bounds__(256)
void cvt_transpose(const float* __restrict__ W, unsigned short* __restrict__ Wt,
                   int K, int N) {
    __shared__ unsigned short tile[32][33];
    int n0 = blockIdx.x * 32, k0 = blockIdx.y * 32;
    int t = threadIdx.x;
#pragma unroll
    for (int i = 0; i < 4; ++i) {
        int idx = i * 256 + t, r = idx >> 5, c = idx & 31;    // r: k-local, c: n-local
        tile[c][r] = bf16of(W[(size_t)(k0 + r) * N + n0 + c]);
    }
    __syncthreads();
#pragma unroll
    for (int i = 0; i < 4; ++i) {
        int idx = i * 256 + t, r = idx >> 5, c = idx & 31;    // r: n-local, c: k-local
        Wt[(size_t)(n0 + r) * K + k0 + c] = tile[r][c];
    }
}

// ---------------------------------------------------------------------------
// Embedding gather: h (fp32) and hb (bf16) copies
// ---------------------------------------------------------------------------
__global__ void embed_kernel(const int* __restrict__ x, const float* __restrict__ emb,
                             float* __restrict__ h, unsigned short* __restrict__ hb,
                             int D, float scl) {
    int row = blockIdx.x;
    int tok = x[row];
    for (int d = threadIdx.x; d < D; d += blockDim.x) {
        float v = emb[(size_t)tok * D + d] * scl;
        h[(size_t)row * D + d]  = v;
        hb[(size_t)row * D + d] = bf16of(v);
    }
}

// ---------------------------------------------------------------------------
// Fused residual-add + LayerNorm (in place on h), dual fp32 + bf16 output
// ---------------------------------------------------------------------------
__global__ __launch_bounds__(256)
void add_ln_kernel(float* __restrict__ h, unsigned short* __restrict__ hb,
                   const float* __restrict__ y,
                   const float* __restrict__ g, const float* __restrict__ bta, int D) {
    __shared__ float ws_[8], ws2_[8];
    int row = blockIdx.x, t = threadIdx.x;
    int wave = t >> 5, lane = t & 31;
    float x[4], s = 0.f, s2 = 0.f;
#pragma unroll
    for (int i = 0; i < 4; ++i) {
        int d = t + i * 256;
        x[i] = h[(size_t)row * D + d] + y[(size_t)row * D + d];
        s += x[i]; s2 += x[i] * x[i];
    }
#pragma unroll
    for (int off = 16; off; off >>= 1) {
        s  += __shfl_xor(s,  off, 32);
        s2 += __shfl_xor(s2, off, 32);
    }
    if (lane == 0) { ws_[wave] = s; ws2_[wave] = s2; }
    __syncthreads();
    s = 0.f; s2 = 0.f;
#pragma unroll
    for (int w = 0; w < 8; ++w) { s += ws_[w]; s2 += ws2_[w]; }
    float mu   = s / (float)D;
    float var  = s2 / (float)D - mu * mu;
    float rstd = rsqrtf(var + 1e-5f);
#pragma unroll
    for (int i = 0; i < 4; ++i) {
        int d = t + i * 256;
        float v = (x[i] - mu) * rstd * g[d] + bta[d];
        h[(size_t)row * D + d]  = v;
        hb[(size_t)row * D + d] = bf16of(v);
    }
}

// ---------------------------------------------------------------------------
// WMMA GEMM: C[M,N] = A_bf16[M,K] x Wt_bf16[N,K]^T + bias
// Block tile 128 x TN, 256 threads = 8 waves (4x2), wave tile 32 x (TN/2).
// Triple-buffered LDS, async global->LDS staging, 1 barrier per K-step.
// blockIdx.z selects among up to 3 fused (Wt, bias, out) problem instances.
// ACT: 0 none, 1 exact GELU, 2 start_pos row mask.  OBF: 1 -> bf16 output.
// ---------------------------------------------------------------------------
struct GemmArgs {
    const unsigned short* Wt[3];
    const float* bias[3];
    void* out[3];
};

template <int ACT, int TN, int OBF>
__global__ __launch_bounds__(256)
void gemm_wmma(const unsigned short* __restrict__ A, GemmArgs ga,
               int M, int N, int K, const int* __restrict__ start_pos, int S) {
    constexpr int NT = TN / 32;          // 16-col tiles per wave
    constexpr int PB = 4 + TN / 32;      // async b64 chunks per thread per batch

    __shared__ unsigned As[3][128][18];  // [buf][row][k-dword], 16 data + 2 pad (72B)
    __shared__ unsigned Bs[3][TN][18];

    const unsigned short* __restrict__ Wt = ga.Wt[blockIdx.z];
    const float* __restrict__ bias = ga.bias[blockIdx.z];
    void* outp = ga.out[blockIdx.z];

    const int t = threadIdx.x;
    const int wave = t >> 5, lane = t & 31;
    const int wr = wave >> 1, wc = wave & 1;
    const int hi = lane >> 4, lo = lane & 15;
    const int m0 = blockIdx.y * 128;
    const int n0 = blockIdx.x * TN;
    const int nk = K >> 5;

    auto issue = [&](int kc, int buf) {
        int k0 = kc << 5;
#pragma unroll
        for (int i = 0; i < 4; ++i) {               // A: 128 rows x 8 chunks
            int idx = i * 256 + t, row = idx >> 3, c = idx & 7;
            stage8(&As[buf][row][2 * c], A + (size_t)(m0 + row) * K + k0 + 4 * c);
        }
#pragma unroll
        for (int i = 0; i < TN / 32; ++i) {         // B: TN rows x 8 chunks
            int idx = i * 256 + t, n = idx >> 3, c = idx & 7;
            stage8(&Bs[buf][n][2 * c], Wt + (size_t)(n0 + n) * K + k0 + 4 * c);
        }
    };

    v8f acc[2][NT];
#pragma unroll
    for (int tr = 0; tr < 2; ++tr)
#pragma unroll
        for (int tc = 0; tc < NT; ++tc) acc[tr][tc] = v8f_zero();

    issue(0, 0);
    for (int kc = 0; kc < nk; ++kc) {
        int buf = kc % 3;
        if (kc + 1 < nk) { issue(kc + 1, (kc + 1) % 3); WAIT_ASYNC(PB); }
        else             { WAIT_ASYNC(0); }
        __syncthreads();

        Frag a[2], b[NT];
#pragma unroll
        for (int tr = 0; tr < 2; ++tr) {
            int row = 32 * wr + 16 * tr + lo;
#pragma unroll
            for (int j = 0; j < 4; ++j) {
                a[tr].u[j]     = As[buf][row][hi * 8 + j];
                a[tr].u[j + 4] = As[buf][row][hi * 8 + 4 + j];
            }
        }
#pragma unroll
        for (int tc = 0; tc < NT; ++tc) {
            int n = (TN / 2) * wc + 16 * tc + lo;
#pragma unroll
            for (int j = 0; j < 8; ++j) b[tc].u[j] = Bs[buf][n][hi * 8 + j];
        }
#pragma unroll
        for (int tr = 0; tr < 2; ++tr)
#pragma unroll
            for (int tc = 0; tc < NT; ++tc)
                acc[tr][tc] = WMMA_BF16(a[tr].b, b[tc].b, acc[tr][tc]);
    }

    // epilogue
#pragma unroll
    for (int tr = 0; tr < 2; ++tr) {
#pragma unroll
        for (int tc = 0; tc < NT; ++tc) {
            int col = n0 + (TN / 2) * wc + 16 * tc + lo;
            float bb = bias[col];
#pragma unroll
            for (int r = 0; r < 8; ++r) {
                int row = m0 + 32 * wr + 16 * tr + r + 8 * hi;
                float v = acc[tr][tc][r] + bb;
                if (ACT == 1)
                    v = 0.5f * v * (1.f + erff(v * 0.70710678118654752f));
                if (ACT == 2) {
                    int b_ = row / S, s_ = row - b_ * S;
                    if (s_ < start_pos[b_]) v = 0.f;
                }
                if (OBF) ((unsigned short*)outp)[(size_t)row * N + col] = bf16of(v);
                else     ((float*)outp)[(size_t)row * N + col] = v;
            }
        }
    }
}

// ---------------------------------------------------------------------------
// Flash attention (ALiBi + tidal mask), bf16 in / bf16 out, online softmax.
// 128 threads = 4 waves; wave owns 16 query rows; 64 keys per iteration:
// 8 WMMA scores + 2 WMMA row-sum (ones trick) + 8 WMMA P*V.
// ---------------------------------------------------------------------------
__global__ __launch_bounds__(128)
void flash_attn(const unsigned short* __restrict__ qb, const unsigned short* __restrict__ kb,
                const unsigned short* __restrict__ vb, const int* __restrict__ start_pos,
                unsigned short* __restrict__ outb, int S, int H, int DK) {
    __shared__ unsigned Kt[64][36];            // [key][d-dword 0..31], +4 pad (144B rows)
    __shared__ unsigned Vt[64][33];            // [d][key-pair 0..31], +1 pad
    __shared__ unsigned short Pst[4][16][66];  // per-wave P staging (bf16)

    const int qt = blockIdx.x;                 // S/64
    const int bh = blockIdx.y;                 // B*H
    const int b = bh / H, h = bh - b * H;
    const int t = threadIdx.x, wave = t >> 5, lane = t & 31;
    const int hi = lane >> 4, lo = lane & 15;
    const int D = H * DK;
    const int sp = start_pos[b];
    const float slope = exp2f(-0.5f * (float)(h + 1));
    const int q0 = qt * 64 + wave * 16;

    // Q fragments (16x64 A-layout, 2 K-chunks), straight bf16 dword loads
    Frag qf[2];
#pragma unroll
    for (int c = 0; c < 2; ++c)
#pragma unroll
        for (int j = 0; j < 8; ++j) {
            int d = c * 32 + hi * 16 + 2 * j;
            qf[c].u[j] = *(const unsigned*)(qb + (size_t)(b * S + q0 + lo) * D + h * DK + d);
        }

    Frag ones;
#pragma unroll
    for (int j = 0; j < 8; ++j) ones.u[j] = 0x3F803F80u;   // bf16 1.0 pair

    v8f o[4];
#pragma unroll
    for (int oc = 0; oc < 4; ++oc) o[oc] = v8f_zero();
    float m_[8], l_[8];
#pragma unroll
    for (int r = 0; r < 8; ++r) { m_[r] = -3.0e38f; l_[r] = 0.f; }

    for (int kt = 0; kt < S; kt += 64) {
        __syncthreads();                       // previous tile fully consumed
        // K tile: 64 keys x 64 d, async b64 chunks (8 per thread)
#pragma unroll
        for (int i = 0; i < 8; ++i) {
            int idx = i * 128 + t, key = idx >> 4, c = idx & 15;
            stage8(&Kt[key][2 * c], kb + (size_t)(b * S + kt + key) * D + h * DK + 4 * c);
        }
        // V tile transposed: [d][key-pair] (VALU pack, 16 per thread)
#pragma unroll
        for (int i = 0; i < 16; ++i) {
            int idx = i * 128 + t, d = idx & 63, pp = idx >> 6;
            unsigned a0 = vb[(size_t)(b * S + kt + 2 * pp)     * D + h * DK + d];
            unsigned a1 = vb[(size_t)(b * S + kt + 2 * pp + 1) * D + h * DK + d];
            Vt[d][pp] = a0 | (a1 << 16);
        }
        WAIT_ASYNC(0);
        __syncthreads();

        // scores: 4 key tiles x 2 K-chunks
        v8f sc[4];
#pragma unroll
        for (int tc = 0; tc < 4; ++tc) sc[tc] = v8f_zero();
#pragma unroll
        for (int tc = 0; tc < 4; ++tc)
#pragma unroll
            for (int c = 0; c < 2; ++c) {
                Frag bf;
#pragma unroll
                for (int j = 0; j < 8; ++j)
                    bf.u[j] = Kt[16 * tc + lo][16 * c + hi * 8 + j];
                sc[tc] = WMMA_BF16(qf[c].b, bf.b, sc[tc]);
            }

        // online softmax: row max via lane shuffles; exp -> LDS (bf16)
        __builtin_amdgcn_wave_barrier();
#pragma unroll
        for (int r = 0; r < 8; ++r) {
            int qrow = q0 + r + 8 * hi;
            float vals[4];
#pragma unroll
            for (int tc = 0; tc < 4; ++tc) {
                int key = kt + 16 * tc + lo;
                float xv = sc[tc][r] * 0.125f + slope * (float)(key - qrow);
                bool keep = (qrow >= key) || (qrow < sp) || (key < sp);
                vals[tc] = keep ? xv : -1.0e9f;
            }
            float tmax = fmaxf(fmaxf(vals[0], vals[1]), fmaxf(vals[2], vals[3]));
#pragma unroll
            for (int off = 1; off < 16; off <<= 1)
                tmax = fmaxf(tmax, __shfl_xor(tmax, off, 32));
            float nm = fmaxf(m_[r], tmax);
            float s = __expf(m_[r] - nm);
            m_[r] = nm;
            l_[r] *= s;
#pragma unroll
            for (int oc = 0; oc < 4; ++oc) o[oc][r] *= s;
#pragma unroll
            for (int tc = 0; tc < 4; ++tc)
                Pst[wave][r + 8 * hi][16 * tc + lo] = bf16of(__expf(vals[tc] - nm));
        }
        __builtin_amdgcn_wave_barrier();

        // P fragments (A layout) for keys 0..31 / 32..63
        Frag pf[2];
#pragma unroll
        for (int kc2 = 0; kc2 < 2; ++kc2)
#pragma unroll
            for (int j = 0; j < 8; ++j) {
                int key = kc2 * 32 + hi * 16 + 2 * j;
                pf[kc2].u[j] = *(const unsigned*)&Pst[wave][lo][key];
            }
        __builtin_amdgcn_wave_barrier();

        // row sums of P via ones-matrix WMMA (replaces shuffle-sum)
        v8f ssum = v8f_zero();
        ssum = WMMA_BF16(pf[0].b, ones.b, ssum);
        ssum = WMMA_BF16(pf[1].b, ones.b, ssum);
#pragma unroll
        for (int r = 0; r < 8; ++r) l_[r] += ssum[r];

        // O += P x V
#pragma unroll
        for (int oc = 0; oc < 4; ++oc)
#pragma unroll
            for (int kc2 = 0; kc2 < 2; ++kc2) {
                Frag vf;
#pragma unroll
                for (int j = 0; j < 8; ++j)
                    vf.u[j] = Vt[16 * oc + lo][kc2 * 16 + hi * 8 + j];
                o[oc] = WMMA_BF16(pf[kc2].b, vf.b, o[oc]);
            }
    }

    // finalize
#pragma unroll
    for (int r = 0; r < 8; ++r) {
        float inv = 1.f / l_[r];
        int qrow = q0 + r + 8 * hi;
#pragma unroll
        for (int oc = 0; oc < 4; ++oc)
            outb[(size_t)(b * S + qrow) * D + h * DK + 16 * oc + lo] =
                bf16of(o[oc][r] * inv);
    }
}

// ---------------------------------------------------------------------------
// Launch: full forward pass
// ---------------------------------------------------------------------------
extern "C" void kernel_launch(void* const* d_in, const int* in_sizes, int n_in,
                              void* d_out, int out_size, void* d_ws, size_t ws_size,
                              hipStream_t stream) {
    (void)in_sizes; (void)n_in; (void)out_size; (void)ws_size;
    const int Bb = 2, S = 1024, D = 1024, H = 16, L = 4, DK = 64, DFF = 4096;
    const int OUTV = 32000;
    const int M = Bb * S;

    const int*   x         = (const int*)  d_in[0];
    const int*   start_pos = (const int*)  d_in[1];
    const float* emb       = (const float*)d_in[2];
    const float* Wq        = (const float*)d_in[3];
    const float* bq        = (const float*)d_in[4];
    const float* Wk        = (const float*)d_in[5];
    const float* bk        = (const float*)d_in[6];
    const float* Wv        = (const float*)d_in[7];
    const float* bv        = (const float*)d_in[8];
    const float* Wo        = (const float*)d_in[9];
    const float* bo        = (const float*)d_in[10];
    const float* ln1_g     = (const float*)d_in[11];
    const float* ln1_b     = (const float*)d_in[12];
    const float* W1        = (const float*)d_in[13];
    const float* b1        = (const float*)d_in[14];
    const float* W2        = (const float*)d_in[15];
    const float* b2        = (const float*)d_in[16];
    const float* ln2_g     = (const float*)d_in[17];
    const float* ln2_b     = (const float*)d_in[18];
    const float* fc_w      = (const float*)d_in[19];
    const float* fc_b      = (const float*)d_in[20];

    // ---- workspace carve-out ----
    char* p = (char*)d_ws;
    auto alloc = [&](size_t bytes) { char* r = p; p += (bytes + 255) & ~(size_t)255; return (void*)r; };

    unsigned short *wtq[4], *wtk[4], *wtv[4], *wto[4], *wt1[4], *wt2[4];
    for (int l = 0; l < L; ++l) {
        wtq[l] = (unsigned short*)alloc((size_t)D * D * 2);
        wtk[l] = (unsigned short*)alloc((size_t)D * D * 2);
        wtv[l] = (unsigned short*)alloc((size_t)D * D * 2);
        wto[l] = (unsigned short*)alloc((size_t)D * D * 2);
        wt1[l] = (unsigned short*)alloc((size_t)D * DFF * 2);
        wt2[l] = (unsigned short*)alloc((size_t)DFF * D * 2);
    }
    unsigned short* wtfc = (unsigned short*)alloc((size_t)D * OUTV * 2);

    float*          h     = (float*)alloc((size_t)M * D * 4);
    unsigned short* hb    = (unsigned short*)alloc((size_t)M * D * 2);
    unsigned short* qbuf  = (unsigned short*)alloc((size_t)M * D * 2);
    unsigned short* kbuf  = (unsigned short*)alloc((size_t)M * D * 2);
    unsigned short* vbuf  = (unsigned short*)alloc((size_t)M * D * 2);
    unsigned short* attnb = (unsigned short*)alloc((size_t)M * D * 2);
    float*          tmp   = (float*)alloc((size_t)M * D * 4);
    unsigned short* ffb   = (unsigned short*)alloc((size_t)M * DFF * 2);

    // ---- weight convert + transpose (once per launch) ----
    for (int l = 0; l < L; ++l) {
        cvt_transpose<<<dim3(D / 32, D / 32), 256, 0, stream>>>(Wq + (size_t)l * D * D, wtq[l], D, D);
        cvt_transpose<<<dim3(D / 32, D / 32), 256, 0, stream>>>(Wk + (size_t)l * D * D, wtk[l], D, D);
        cvt_transpose<<<dim3(D / 32, D / 32), 256, 0, stream>>>(Wv + (size_t)l * D * D, wtv[l], D, D);
        cvt_transpose<<<dim3(D / 32, D / 32), 256, 0, stream>>>(Wo + (size_t)l * D * D, wto[l], D, D);
        cvt_transpose<<<dim3(DFF / 32, D / 32), 256, 0, stream>>>(W1 + (size_t)l * D * DFF, wt1[l], D, DFF);
        cvt_transpose<<<dim3(D / 32, DFF / 32), 256, 0, stream>>>(W2 + (size_t)l * DFF * D, wt2[l], DFF, D);
    }
    cvt_transpose<<<dim3(OUTV / 32, D / 32), 256, 0, stream>>>(fc_w, wtfc, D, OUTV);

    embed_kernel<<<M, 256, 0, stream>>>(x, emb, h, hb, D, 32.0f /* sqrt(1024) */);

    dim3 blk(256);
    for (int l = 0; l < L; ++l) {
        const size_t oD = (size_t)l * D, oF = (size_t)l * DFF;

        GemmArgs qkv;
        qkv.Wt[0] = wtq[l];  qkv.Wt[1] = wtk[l];  qkv.Wt[2] = wtv[l];
        qkv.bias[0] = bq + oD; qkv.bias[1] = bk + oD; qkv.bias[2] = bv + oD;
        qkv.out[0] = qbuf; qkv.out[1] = kbuf; qkv.out[2] = vbuf;
        gemm_wmma<0, 64, 1><<<dim3(D / 64, M / 128, 3), blk, 0, stream>>>(
            hb, qkv, M, D, D, nullptr, S);

        flash_attn<<<dim3(S / 64, Bb * H), dim3(128), 0, stream>>>(
            qbuf, kbuf, vbuf, start_pos, attnb, S, H, DK);

        GemmArgs go{};
        go.Wt[0] = wto[l]; go.bias[0] = bo + oD; go.out[0] = tmp;
        gemm_wmma<0, 64, 0><<<dim3(D / 64, M / 128, 1), blk, 0, stream>>>(
            attnb, go, M, D, D, nullptr, S);
        add_ln_kernel<<<M, 256, 0, stream>>>(h, hb, tmp, ln1_g + oD, ln1_b + oD, D);

        GemmArgs g1{};
        g1.Wt[0] = wt1[l]; g1.bias[0] = b1 + oF; g1.out[0] = ffb;
        gemm_wmma<1, 128, 1><<<dim3(DFF / 128, M / 128, 1), blk, 0, stream>>>(
            hb, g1, M, DFF, D, nullptr, S);

        GemmArgs g2{};
        g2.Wt[0] = wt2[l]; g2.bias[0] = b2 + oD; g2.out[0] = tmp;
        gemm_wmma<0, 64, 0><<<dim3(D / 64, M / 128, 1), blk, 0, stream>>>(
            ffb, g2, M, D, DFF, nullptr, S);
        add_ln_kernel<<<M, 256, 0, stream>>>(h, hb, tmp, ln2_g + oD, ln2_b + oD, D);
    }

    GemmArgs gf{};
    gf.Wt[0] = wtfc; gf.bias[0] = fc_b; gf.out[0] = d_out;
    gemm_wmma<2, 128, 0><<<dim3(OUTV / 128, M / 128, 1), blk, 0, stream>>>(
        hb, gf, M, OUTV, D, start_pos, S);
}
